// PrototypeDist_54099408060818
// MI455X (gfx1250) — matrix operand order repaired and checked by reading, not compiled, not run on hardware
//
#include <hip/hip_runtime.h>

typedef __bf16 v16bf __attribute__((ext_vector_type(16)));
typedef __bf16 v8bf  __attribute__((ext_vector_type(8)));
typedef __bf16 v4bf  __attribute__((ext_vector_type(4)));
typedef float  v8f   __attribute__((ext_vector_type(8)));
typedef float  f4    __attribute__((ext_vector_type(4)));

#define NB 16384   // batch rows
#define NP 4096    // prototypes
#define ND 64      // feature dim

// ---------------------------------------------------------------------------
// Kernel 1: row norms  xsq[b] = sum_d x[b,d]^2 ; wsq[p] = sum_d W[p,d]^2
// ---------------------------------------------------------------------------
__global__ void pd_norms(const float* __restrict__ x, const float* __restrict__ W,
                         float* __restrict__ xsq, float* __restrict__ wsq) {
    int t = blockIdx.x * 256 + threadIdx.x;
    const float* src;
    float* dst;
    if (t < NB) {
        src = x + (size_t)t * ND;
        dst = xsq + t;
    } else if (t < NB + NP) {
        int p = t - NB;
        src = W + (size_t)p * ND;
        dst = wsq + p;
    } else {
        return;
    }
    float s = 0.0f;
#pragma unroll
    for (int j = 0; j < ND; j += 4) {
        f4 v = *(const f4*)(src + j);
        s += v.x * v.x + v.y * v.y + v.z * v.z + v.w * v.w;
    }
    *dst = s;
}

// ---------------------------------------------------------------------------
// Kernel 2: split f32 -> bf16 hi/lo.
//   x  -> xhi/xlo, same [NB, ND] row-major (WMMA A frags: lane=M, elems=K,
//         contiguous along K).
//   W  -> whiT/wloT transposed to [ND, NP] so WMMA B frags (lane=N, elems=K)
//         read 16 consecutive N (32 bytes) across lanes for each K.
// ---------------------------------------------------------------------------
__global__ void pd_split(const float* __restrict__ x, const float* __restrict__ W,
                         __bf16* __restrict__ xhi, __bf16* __restrict__ xlo,
                         __bf16* __restrict__ whiT, __bf16* __restrict__ wloT) {
    int t = blockIdx.x * 256 + threadIdx.x;
    const int NX4 = NB * ND / 4;
    const int NW4 = NP * ND / 4;
    if (t < NX4) {
        f4 v = *(const f4*)(x + (size_t)t * 4);
        v4bf h, l;
#pragma unroll
        for (int c = 0; c < 4; ++c) {
            float f = v[c];
            __bf16 hb = (__bf16)f;           // RNE f32->bf16
            h[c] = hb;
            l[c] = (__bf16)(f - (float)hb);  // residual
        }
        *(v4bf*)(xhi + (size_t)t * 4) = h;
        *(v4bf*)(xlo + (size_t)t * 4) = l;
    } else if (t < NX4 + NW4) {
        int i = (t - NX4) * 4;   // flat index into W [NP, ND]
        int p = i / ND;
        int d = i % ND;          // 4 consecutive d values
        f4 v = *(const f4*)(W + i);
#pragma unroll
        for (int c = 0; c < 4; ++c) {
            float f = v[c];
            __bf16 hb = (__bf16)f;
            whiT[(size_t)(d + c) * NP + p] = hb;
            wloT[(size_t)(d + c) * NP + p] = (__bf16)(f - (float)hb);
        }
    }
}

// ---------------------------------------------------------------------------
// Kernel 3: tiled distance GEMM on v_wmma_f32_16x16x32_bf16.
//   Workgroup: 256 threads (8 waves), tile 128(M) x 64(N).
//   Wave: 32x32 tile = 2x2 WMMA tiles; K=64 unrolled (2 chunks of 32).
//   3-term bf16 split per chunk: hi*hi + hi*lo + lo*hi  (~fp32 accuracy).
//   Epilogue: out = xsq + wsq - 2*acc.
// ---------------------------------------------------------------------------
__global__ __launch_bounds__(256, 1) void pd_wmma(
    const __bf16* __restrict__ xhi, const __bf16* __restrict__ xlo,
    const __bf16* __restrict__ whiT, const __bf16* __restrict__ wloT,
    const float* __restrict__ xsq, const float* __restrict__ wsq,
    float* __restrict__ out) {
    const int lane = threadIdx.x & 31;
    const int wave = threadIdx.x >> 5;
    const int half = lane >> 4;   // 0: lanes 0-15, 1: lanes 16-31
    const int m16  = lane & 15;

    const int blockM = blockIdx.y * 128;  // batch rows
    const int blockN = blockIdx.x * 64;   // prototype cols
    const int wi = wave & 3;              // M sub-block (4)
    const int wj = wave >> 2;             // N sub-block (2)
    const int waveM = blockM + wi * 32;
    const int waveN = blockN + wj * 32;

    v8f acc[2][2];
#pragma unroll
    for (int a = 0; a < 2; ++a)
#pragma unroll
        for (int b = 0; b < 2; ++b)
            acc[a][b] = (v8f){0.f, 0.f, 0.f, 0.f, 0.f, 0.f, 0.f, 0.f};

#pragma unroll
    for (int kb = 0; kb < ND; kb += 32) {
        // ---- A fragments (x): lane m16 = row; elem j<8 -> K=kb+half*8+j,
        //                        elem j>=8 -> K=kb+16+half*8+(j-8)
        v16bf ah[2], al[2];
#pragma unroll
        for (int ti = 0; ti < 2; ++ti) {
            int row = waveM + ti * 16 + m16;
            const __bf16* ph = xhi + (size_t)row * ND + kb + half * 8;
            const __bf16* pl = xlo + (size_t)row * ND + kb + half * 8;
            v8bf h0 = *(const v8bf*)(ph);
            v8bf h1 = *(const v8bf*)(ph + 16);
            v8bf l0 = *(const v8bf*)(pl);
            v8bf l1 = *(const v8bf*)(pl + 16);
#pragma unroll
            for (int e = 0; e < 8; ++e) {
                ah[ti][e] = h0[e]; ah[ti][e + 8] = h1[e];
                al[ti][e] = l0[e]; al[ti][e + 8] = l1[e];
            }
        }
        // ---- B fragments (W^T): lane m16 = column N; elem j<8 -> K=kb+half*8+j,
        //                        elem j>=8 -> K=kb+16+half*8+(j-8)
        v16bf bh[2], bl[2];
#pragma unroll
        for (int tj = 0; tj < 2; ++tj) {
            int col = waveN + tj * 16 + m16;
            const __bf16* ph = whiT + (size_t)(kb + half * 8) * NP + col;
            const __bf16* pl = wloT + (size_t)(kb + half * 8) * NP + col;
#pragma unroll
            for (int e = 0; e < 8; ++e) {
                bh[tj][e] = ph[(size_t)e * NP]; bh[tj][e + 8] = ph[(size_t)(16 + e) * NP];
                bl[tj][e] = pl[(size_t)e * NP]; bl[tj][e + 8] = pl[(size_t)(16 + e) * NP];
            }
        }
        // ---- 2x2 tiles x 3 split-terms = 12 WMMAs per K-chunk
#pragma unroll
        for (int ti = 0; ti < 2; ++ti)
#pragma unroll
            for (int tj = 0; tj < 2; ++tj) {
                acc[ti][tj] = __builtin_amdgcn_wmma_f32_16x16x32_bf16(
                    false, ah[ti], false, bh[tj], (short)0, acc[ti][tj], false, false);
                acc[ti][tj] = __builtin_amdgcn_wmma_f32_16x16x32_bf16(
                    false, ah[ti], false, bl[tj], (short)0, acc[ti][tj], false, false);
                acc[ti][tj] = __builtin_amdgcn_wmma_f32_16x16x32_bf16(
                    false, al[ti], false, bh[tj], (short)0, acc[ti][tj], false, false);
            }
    }

    // ---- Epilogue: C layout lane(0-15)=N, VGPR r -> M = r + 8*half
#pragma unroll
    for (int ti = 0; ti < 2; ++ti) {
        int rowBase = waveM + ti * 16 + half * 8;
        float xs[8];
#pragma unroll
        for (int r = 0; r < 8; ++r) xs[r] = xsq[rowBase + r];
#pragma unroll
        for (int tj = 0; tj < 2; ++tj) {
            int col = waveN + tj * 16 + m16;
            float wn = wsq[col];
#pragma unroll
            for (int r = 0; r < 8; ++r) {
                size_t idx = (size_t)(rowBase + r) * NP + col;
                out[idx] = fmaf(-2.0f, acc[ti][tj][r], xs[r] + wn);
            }
        }
    }
}

// ---------------------------------------------------------------------------
extern "C" void kernel_launch(void* const* d_in, const int* in_sizes, int n_in,
                              void* d_out, int out_size, void* d_ws, size_t ws_size,
                              hipStream_t stream) {
    (void)in_sizes; (void)n_in; (void)out_size; (void)ws_size;
    const float* x = (const float*)d_in[0];
    const float* W = (const float*)d_in[1];   // [1, NP, ND] -> flat [NP, ND]
    float* out = (float*)d_out;

    // Workspace layout (~5.1 MB total):
    char* ws = (char*)d_ws;
    float*  xsq  = (float*)(ws);                                   // 64 KB
    float*  wsq  = (float*)(ws + 65536);                           // 16 KB
    __bf16* xhi  = (__bf16*)(ws + 81920);                          // 2 MB
    __bf16* xlo  = (__bf16*)(ws + 81920 + (size_t)NB * ND * 2);    // 2 MB
    __bf16* whiT = (__bf16*)(ws + 81920 + 2 * (size_t)NB * ND * 2);            // 512 KB
    __bf16* wloT = (__bf16*)(ws + 81920 + 2 * (size_t)NB * ND * 2
                                         + (size_t)NP * ND * 2);               // 512 KB

    int totN = NB + NP;
    pd_norms<<<(totN + 255) / 256, 256, 0, stream>>>(x, W, xsq, wsq);

    int totS = (NB * ND + NP * ND) / 4;
    pd_split<<<(totS + 255) / 256, 256, 0, stream>>>(x, W, xhi, xlo, whiT, wloT);

    dim3 grid(NP / 64, NB / 128);
    pd_wmma<<<grid, 256, 0, stream>>>(xhi, xlo, whiT, wloT, xsq, wsq, out);
}
